// Embedding_CRF_6554120093704
// MI455X (gfx1250) — matrix operand order repaired and checked by reading, not compiled, not run on hardware
//
#include <hip/hip_runtime.h>
#include <hip/hip_bf16.h>

typedef __attribute__((ext_vector_type(16))) _Float16 v16h;
typedef __attribute__((ext_vector_type(8)))  float    v8f;

#define NUM_LABELS 64
#define SEQ_L      1024
#define BATCH      128
#define DIM        256
#define T_START    0
#define T_END      1

// ---------------------------------------------------------------------------
// Kernel 0: zero the scalar output (harness poisons d_out; we must re-init
// every call so graph replays are deterministic).
// ---------------------------------------------------------------------------
__global__ void zero_out_kernel(float* out) {
  if (threadIdx.x == 0) out[0] = 0.0f;
}

// ---------------------------------------------------------------------------
// Kernel A: emissions = embed[x] @ W + b   (f16 WMMA, f32 accumulate)
// Each block: 64 tokens x 64 labels, K=256. 8 waves; wave w computes a
// 16x32 output slab (two 16x16 WMMA tiles), K-loop of 8 x 32.
// LDS: A tile 64x256 f16 (32KB) + W transposed 64x256 f16 (32KB).
// ---------------------------------------------------------------------------
__global__ __launch_bounds__(256) void emissions_wmma_kernel(
    const int*   __restrict__ x,
    const float* __restrict__ embed,
    const float* __restrict__ W,      // [DIM, NUM_LABELS] row-major
    const float* __restrict__ bias,   // [NUM_LABELS]
    float*       __restrict__ emis)   // [B*L, NUM_LABELS]
{
  __shared__ _Float16 Ah[64][DIM];          // gathered tokens, f16
  __shared__ _Float16 Wh[NUM_LABELS][DIM];  // W transposed: [label][dim], f16

  const int  t    = threadIdx.x;
  const long tok0 = (long)blockIdx.x * 64;

  // --- gather 64 embedding rows, convert f32 -> f16 into LDS (coalesced) ---
  {
    const int  r   = t >> 2;          // token row 0..63
    const int  seg = (t & 3) * 64;    // 64-col segment
    const long row = (long)x[tok0 + r];
    const float4* src = reinterpret_cast<const float4*>(embed + row * DIM + seg);
#pragma unroll
    for (int i = 0; i < 16; ++i) {
      float4 v = src[i];
      int c = seg + i * 4;
      Ah[r][c + 0] = (_Float16)v.x;  Ah[r][c + 1] = (_Float16)v.y;
      Ah[r][c + 2] = (_Float16)v.z;  Ah[r][c + 3] = (_Float16)v.w;
    }
  }
  // --- load W transposed into LDS: Wh[n][k] = W[k*64+n] ---
  {
    const int n  = t & 63;
    const int kb = t >> 6;            // 0..3
#pragma unroll 4
    for (int kk = 0; kk < 64; ++kk) {
      int k = kb * 64 + kk;
      Wh[n][k] = (_Float16)W[k * NUM_LABELS + n];
    }
  }
  __syncthreads();

  const int wave = t >> 5;            // 0..7
  const int lane = t & 31;
  const int hi   = lane >> 4;         // lane half: selects K sub-range
  const int m16  = lane & 15;
  const int R0   = (wave >> 1) * 16;  // token row group
  const int C0   = (wave & 1)  * 32;  // label col group (two 16-wide tiles)

  v8f acc0 = {}; v8f acc1 = {};
#pragma unroll
  for (int kb = 0; kb < 8; ++kb) {
    // A fragment (contiguous k order, same as B):
    //  lanes 0-15: K = 0..15; lanes 16-31: K = 16..31
    const int abase = kb * 32 + hi * 16;
    v16h a;
#pragma unroll
    for (int j = 0; j < 8; ++j) a[j]     = Ah[R0 + m16][abase + j];
#pragma unroll
    for (int j = 0; j < 8; ++j) a[8 + j] = Ah[R0 + m16][abase + 8 + j];

    // B fragments (same contiguous k order as A): lanes 0-15 hold K 0..15, lanes 16-31 K 16..31
    const int kbase = kb * 32 + hi * 16;
    v16h b0, b1;
#pragma unroll
    for (int j = 0; j < 16; ++j) b0[j] = Wh[C0 + m16][kbase + j];
#pragma unroll
    for (int j = 0; j < 16; ++j) b1[j] = Wh[C0 + 16 + m16][kbase + j];

    acc0 = __builtin_amdgcn_wmma_f32_16x16x32_f16(false, a, false, b0,
                                                  (short)0, acc0, false, false);
    acc1 = __builtin_amdgcn_wmma_f32_16x16x32_f16(false, a, false, b1,
                                                  (short)0, acc1, false, false);
  }

  const float bv0 = bias[C0 + m16];
  const float bv1 = bias[C0 + 16 + m16];
  // D layout: lane<16 holds rows M=v, lane>=16 rows M=v+8; column = lane&15
#pragma unroll
  for (int v = 0; v < 8; ++v) {
    long tok = tok0 + R0 + v + hi * 8;
    emis[tok * NUM_LABELS + C0 + m16]      = acc0[v] + bv0;
    emis[tok * NUM_LABELS + C0 + 16 + m16] = acc1[v] + bv1;
  }
}

// ---------------------------------------------------------------------------
// Kernel B: CRF forward recursion + gold score, one block per batch.
// 64 threads (2 waves), thread j owns label j and holds column j of
// exp(trans) in 64 VGPRs. Exp-domain recursion with rescale every 4 steps.
// ---------------------------------------------------------------------------
__global__ __launch_bounds__(64) void crf_forward_kernel(
    const float* __restrict__ emis,
    const int*   __restrict__ tags,
    const float* __restrict__ trans,
    float*       __restrict__ out)
{
  __shared__ __align__(16) float abuf[2][NUM_LABELS]; // double-buffered alpha
  __shared__ float red[NUM_LABELS];
  __shared__ float wm[2];

  const int b    = blockIdx.x;
  const int j    = threadIdx.x;
  const int lane = j & 31;
  const int wv   = j >> 5;

  // column j of exp(trans): exp(-10000) -> 0 exactly (log-semiring zero)
  float Et[NUM_LABELS];
#pragma unroll 8
  for (int i = 0; i < NUM_LABELS; ++i) Et[i] = __expf(trans[i * NUM_LABELS + j]);

  const float* eb = emis + (long)b * SEQ_L * NUM_LABELS;

  // init: alpha0 = trans[START] + emissions[:,0]; shift by block max
  float nl = trans[T_START * NUM_LABELS + j] + eb[j];
  float m = nl;
#pragma unroll
  for (int d = 16; d > 0; d >>= 1) m = fmaxf(m, __shfl_xor(m, d, 32));
  if (lane == 0) wm[wv] = m;
  __syncthreads();
  m = fmaxf(wm[0], wm[1]);
  float logScale = m;
  abuf[0][j] = __expf(nl - m);
  __syncthreads();

  int cur = 0;
  for (int ts = 1; ts < SEQ_L; ++ts) {
    float e = eb[ts * NUM_LABELS + j];
    if (ts + 1 < SEQ_L)
      __builtin_prefetch(eb + (ts + 1) * NUM_LABELS + j, 0, 0);

    // s_j = sum_i a_i * expTrans[i][j]  (LDS broadcast, fully unrolled FMAs)
    const float4* ap = reinterpret_cast<const float4*>(&abuf[cur][0]);
    float s = 0.0f;
#pragma unroll
    for (int i4 = 0; i4 < 16; ++i4) {
      float4 av = ap[i4];
      s += av.x * Et[4 * i4 + 0];
      s += av.y * Et[4 * i4 + 1];
      s += av.z * Et[4 * i4 + 2];
      s += av.w * Et[4 * i4 + 3];
    }
    float v = s * __expf(e);

    if ((ts & 3) == 3) {               // rescale every 4 steps (fp32 range)
      float mm = v;
#pragma unroll
      for (int d = 16; d > 0; d >>= 1) mm = fmaxf(mm, __shfl_xor(mm, d, 32));
      if (lane == 0) wm[wv] = mm;
      __syncthreads();
      mm = fmaxf(wm[0], wm[1]);
      logScale += __logf(mm);
      v *= __builtin_amdgcn_rcpf(mm);
    }

    abuf[cur ^ 1][j] = v;
    __syncthreads();                   // single barrier per step
    cur ^= 1;
  }

  // log Z = logScale + log( sum_j a_j * exp(trans[j][END]) )
  float term = abuf[cur][j] * __expf(trans[j * NUM_LABELS + T_END]);
  red[j] = term;
  __syncthreads();
  if (j < 32) red[j] += red[j + 32];
  __syncthreads();
  float ssum = 0.0f;
  if (j < 32) {
    float r = red[j];
#pragma unroll
    for (int d = 16; d > 0; d >>= 1) r += __shfl_xor(r, d, 32);
    ssum = r;
  }
  __syncthreads();                     // red reuse (WAR)

  // gold path score (mask is all-ones in reference)
  const int* tg = tags + (long)b * SEQ_L;
  float sc = 0.0f;
  for (int ts = j; ts < SEQ_L; ts += NUM_LABELS) {
    int tt = tg[ts];
    sc += eb[ts * NUM_LABELS + tt];
    if (ts + 1 < SEQ_L) sc += trans[tt * NUM_LABELS + tg[ts + 1]];
  }
  if (j == 0) sc += trans[T_START * NUM_LABELS + tg[0]];
  if (j == 1) sc += trans[tg[SEQ_L - 1] * NUM_LABELS + T_END];
  red[j] = sc;
  __syncthreads();
  if (j < 32) red[j] += red[j + 32];
  __syncthreads();
  if (j < 32) {
    float r = red[j];
#pragma unroll
    for (int d = 16; d > 0; d >>= 1) r += __shfl_xor(r, d, 32);
    if (j == 0) {
      float logZ = logScale + __logf(ssum);
      atomicAdd(out, logZ - r);        // nll contribution of this sequence
    }
  }
}

// ---------------------------------------------------------------------------
extern "C" void kernel_launch(void* const* d_in, const int* in_sizes, int n_in,
                              void* d_out, int out_size, void* d_ws, size_t ws_size,
                              hipStream_t stream) {
  const int*   x     = (const int*)d_in[0];
  const int*   tags  = (const int*)d_in[1];
  // d_in[2] = mask (ignored by reference CRF)
  const float* embed = (const float*)d_in[3];
  const float* W     = (const float*)d_in[4];
  const float* bias  = (const float*)d_in[5];
  const float* trans = (const float*)d_in[6];
  float*       out   = (float*)d_out;
  float*       emis  = (float*)d_ws;   // B*L*64 f32 = 32 MiB

  zero_out_kernel<<<1, 32, 0, stream>>>(out);
  emissions_wmma_kernel<<<(BATCH * SEQ_L) / 64, 256, 0, stream>>>(
      x, embed, W, bias, emis);
  crf_forward_kernel<<<BATCH, 64, 0, stream>>>(emis, tags, trans, out);
}
